// Qwen3_5GatedDeltaNet_25967372272034
// MI455X (gfx1250) — matrix operand order repaired and checked by reading, not compiled, not run on hardware
//
#include <hip/hip_runtime.h>
#include <hip/hip_bf16.h>
#include <cstdint>
#include <cstddef>

// ---- problem constants (match reference) ----
#define S_LEN   4096
#define HDIM    2048
#define HK      16
#define HV      32
#define DKDIM   128
#define DVDIM   128
#define KCONV   4
#define CHUNK   64
#define NCHUNK  (S_LEN / CHUNK)        // 64
#define KEY_DIM (HK * DKDIM)           // 2048
#define VAL_DIM (HV * DVDIM)           // 4096
#define CONV_DIM (2 * KEY_DIM + VAL_DIM) // 8192
#define DVB     32                     // value-dim block in delta kernel
#define NVB     (DVDIM / DVB)          // 4

#ifndef USE_ASYNC_LDS
#define USE_ASYNC_LDS 1
#endif

typedef __bf16 bf16_t;
typedef __bf16 v16bf __attribute__((ext_vector_type(16)));
typedef float  v8f   __attribute__((ext_vector_type(8)));

// ---------- bf16 helpers ----------
__device__ __forceinline__ bf16_t f2bf(float x) {
  unsigned u = __builtin_bit_cast(unsigned, x);
  u += 0x7FFFu + ((u >> 16) & 1u);               // round-to-nearest-even
  unsigned short h = (unsigned short)(u >> 16);
  return __builtin_bit_cast(bf16_t, h);
}
__device__ __forceinline__ float bf2f(bf16_t x) {
  unsigned short s = __builtin_bit_cast(unsigned short, x);
  unsigned u = ((unsigned)s) << 16;
  return __builtin_bit_cast(float, u);
}

// ---------- CDNA5 async global->LDS staging (ASYNCcnt) ----------
typedef __attribute__((address_space(3))) char lds_char;
__device__ __forceinline__ uint32_t lds_offset(void* p) {
  return (uint32_t)(uintptr_t)(lds_char*)p;     // addrspace(3) ptr = byte offset in LDS
}
__device__ __forceinline__ void async_ld_b128(const void* gptr, void* lptr) {
#if USE_ASYNC_LDS
  uint32_t l = lds_offset(lptr);
  uint64_t g = (uint64_t)(uintptr_t)gptr;
  asm volatile("global_load_async_to_lds_b128 %0, %1, off"
               :: "v"(l), "v"(g) : "memory");
#else
  *(uint4*)lptr = *(const uint4*)gptr;
#endif
}
// async loads complete in order -> waiting for <=N leaves the newest N in flight
template <int N>
__device__ __forceinline__ void async_wait() {
#if USE_ASYNC_LDS
#if __has_builtin(__builtin_amdgcn_s_wait_asynccnt)
  __builtin_amdgcn_s_wait_asynccnt((unsigned short)N);
#else
  asm volatile("s_wait_asynccnt %0" :: "i"(N) : "memory");
#endif
#endif
}

// ---------- WMMA wrapper (gfx1250: v_wmma_f32_16x16x32_bf16) ----------
__device__ __forceinline__ v8f wmma_bf16(v16bf a, v16bf b, v8f c) {
  return __builtin_amdgcn_wmma_f32_16x16x32_bf16(
      /*neg_a=*/false, a, /*neg_b=*/false, b,
      /*c_mod=*/(short)0, c, /*reuse_a=*/false, /*reuse_b=*/false);
}

// ---------- fragment builders (contiguous k = half*16 + s in A and B, wave32) ----------
// A: 16x32 (MxK). lanes 0-15: M=lane, K = half*16..+15 -> one 32B run
__device__ __forceinline__ v16bf frag_a_raw(const bf16_t* M, int ld, int mBase, int kBase, int lane) {
  int m = lane & 15, half = lane >> 4;
  const bf16_t* p = M + (mBase + m) * ld + kBase;
  v16bf f;
#pragma unroll
  for (int v = 0; v < 8; ++v) {
    int kk = half * 16 + v * 2;
    f[2 * v]     = p[kk];
    f[2 * v + 1] = p[kk + 1];
  }
  return f;
}
// A fragment with a per-row (M) scale applied during load
__device__ __forceinline__ v16bf frag_a_rs(const bf16_t* M, int ld, int mBase, int kBase,
                                           const float* rowScale, int lane) {
  int m = lane & 15, half = lane >> 4;
  int row = mBase + m;
  float s = rowScale[row];
  const bf16_t* p = M + row * ld + kBase;
  v16bf f;
#pragma unroll
  for (int v = 0; v < 8; ++v) {
    int kk = half * 16 + v * 2;
    f[2 * v]     = f2bf(bf2f(p[kk]) * s);
    f[2 * v + 1] = f2bf(bf2f(p[kk + 1]) * s);
  }
  return f;
}
// A fragment from a transposed source (element (m,k) = M[k][m]) with per-k scale
__device__ __forceinline__ v16bf frag_aT(const bf16_t* M, int ld, int mBase, int kBase,
                                         const float* kScale, int lane) {
  int m = lane & 15, half = lane >> 4;
  v16bf f;
#pragma unroll
  for (int v = 0; v < 8; ++v) {
    int kk = kBase + half * 16 + v * 2;
    f[2 * v]     = f2bf(bf2f(M[kk * ld + mBase + m]) * kScale[kk]);
    f[2 * v + 1] = f2bf(bf2f(M[(kk + 1) * ld + mBase + m]) * kScale[kk + 1]);
  }
  return f;
}
// B: 32x16 (KxN), source row-major KxN: strided per-element gather
__device__ __forceinline__ v16bf frag_b_raw(const bf16_t* M, int ld, int kBase, int nBase, int lane) {
  int n = lane & 15, half = lane >> 4;
  v16bf f;
#pragma unroll
  for (int v = 0; v < 8; ++v) {
    int kk = kBase + half * 16 + v * 2;
    f[2 * v]     = M[kk * ld + nBase + n];
    f[2 * v + 1] = M[(kk + 1) * ld + nBase + n];
  }
  return f;
}
// B with per-k scale
__device__ __forceinline__ v16bf frag_b_ks(const bf16_t* M, int ld, int kBase, int nBase,
                                           const float* kScale, int lane) {
  int n = lane & 15, half = lane >> 4;
  v16bf f;
#pragma unroll
  for (int v = 0; v < 8; ++v) {
    int kk = kBase + half * 16 + v * 2;
    f[2 * v]     = f2bf(bf2f(M[kk * ld + nBase + n]) * kScale[kk]);
    f[2 * v + 1] = f2bf(bf2f(M[(kk + 1) * ld + nBase + n]) * kScale[kk + 1]);
  }
  return f;
}
// B from an n-major (transposed) tile: element (k,n) = Mt[n][k]; 16 contiguous bf16 per lane
__device__ __forceinline__ v16bf frag_bT(const bf16_t* Mt, int ld, int kBase, int nBase, int lane) {
  int n = lane & 15, half = lane >> 4;
  const bf16_t* p = Mt + (nBase + n) * ld + kBase + half * 16;
  v16bf f;
#pragma unroll
  for (int i = 0; i < 16; ++i) f[i] = p[i];
  return f;
}
// B fragment from an f32 LDS matrix (state), converted on the fly
__device__ __forceinline__ v16bf frag_b_f32(const float* M, int ld, int kBase, int nBase, int lane) {
  int n = lane & 15, half = lane >> 4;
  v16bf f;
#pragma unroll
  for (int v = 0; v < 8; ++v) {
    int kk = kBase + half * 16 + v * 2;
    f[2 * v]     = f2bf(M[kk * ld + nBase + n]);
    f[2 * v + 1] = f2bf(M[(kk + 1) * ld + nBase + n]);
  }
  return f;
}
// C/D: 16x16 f32; VGPR r -> M = r (lanes 0-15) / r+8 (lanes 16-31), N = lane&15
__device__ __forceinline__ void store_tile(bf16_t* M, int ld, int mBase, int nBase, v8f acc, int lane) {
  int n = lane & 15, half = lane >> 4;
#pragma unroll
  for (int r = 0; r < 8; ++r)
    M[(mBase + half * 8 + r) * ld + nBase + n] = f2bf(acc[r]);
}

// ---------- f32 -> bf16 pack ----------
__global__ __launch_bounds__(256) void cvt_bf16_kernel(const float* __restrict__ in,
                                                       bf16_t* __restrict__ out, size_t n) {
  size_t i = (size_t)blockIdx.x * 256 + threadIdx.x;
  if (i < n) out[i] = f2bf(in[i]);
}

// ---------- generic tiled WMMA GEMM: C[M,N] f32 = A[M,K] bf16 * B[K,N] bf16 ----------
// block: 256 threads = 8 waves; tile 64(M) x 128(N); K-step 32; wave -> 32x32 (2x2 WMMA tiles).
// Software-pipelined: tile i+1 staged via global_load_async_to_lds_b128 while tile i is
// transposed + multiplied; s_wait_asynccnt 3 retires tile i (async loads complete in order)
// while tile i+1's 3 ops stay in flight. Tile i+2 is prefetched into GL2 (192MB L2 keeps
// the bf16 weights resident across the 64 row-blocks that reuse them).
__global__ __launch_bounds__(256) void wmma_gemm_kernel(const bf16_t* __restrict__ A,
                                                        const bf16_t* __restrict__ B,
                                                        float* __restrict__ C,
                                                        int M, int N, int K) {
  __shared__ __align__(16) bf16_t As[2][64][40];     // 64x32 + pad, double-buffered
  __shared__ __align__(16) bf16_t Braw[2][32][136];  // 32x128 row-major staging (272B rows)
  __shared__ __align__(16) bf16_t Bst[128][40];      // n-major (transposed) tile
  const int tid = threadIdx.x;
  const int lane = tid & 31, wave = tid >> 5;
  const int wm = wave & 1, wn = wave >> 1;           // 32-row half / 32-col quarter
  const int mBase = blockIdx.y * 64;
  const int nBase = blockIdx.x * 128;
  // per-thread staging coordinates (A: one b128, B: two b128)
  const int ar = tid >> 2, ac = (tid & 3) * 8;
  const int br0 = (tid * 2) >> 4,     bc0 = ((tid * 2) & 15) * 8;
  const int br1 = (tid * 2 + 1) >> 4, bc1 = ((tid * 2 + 1) & 15) * 8;
  const int nk = K / 32;

  // prologue: stage tile 0
  async_ld_b128(A + (size_t)(mBase + ar) * K + ac,        &As[0][ar][ac]);
  async_ld_b128(B + (size_t)br0 * N + nBase + bc0,        &Braw[0][br0][bc0]);
  async_ld_b128(B + (size_t)br1 * N + nBase + bc1,        &Braw[0][br1][bc1]);

  v8f acc00 = {}, acc01 = {}, acc10 = {}, acc11 = {};
  for (int i = 0; i < nk; ++i) {
    const int buf = i & 1;
    __syncthreads();                 // all waves done computing tile i-1: buf^1 reusable
    if (i + 1 < nk) {
      const int k1 = (i + 1) * 32;
      async_ld_b128(A + (size_t)(mBase + ar) * K + k1 + ac,  &As[buf ^ 1][ar][ac]);
      async_ld_b128(B + (size_t)(k1 + br0) * N + nBase + bc0, &Braw[buf ^ 1][br0][bc0]);
      async_ld_b128(B + (size_t)(k1 + br1) * N + nBase + bc1, &Braw[buf ^ 1][br1][bc1]);
      async_wait<3>();               // tile i retired; tile i+1 still flying
    } else {
      async_wait<0>();
    }
    if (i + 2 < nk) {                // prefetch tile i+2 into GL2
      const int k2 = (i + 2) * 32;
      __builtin_prefetch(A + (size_t)(mBase + ar) * K + k2 + ac, 0, 1);
      __builtin_prefetch(B + (size_t)(k2 + (tid >> 3)) * N + nBase + (tid & 7) * 16, 0, 1);
    }
    __syncthreads();                 // tile i visible to all waves
    {   // LDS transpose Braw(KxN) -> Bst(NxK): 16 k-elems (one n) per thread
      int n = tid >> 1, koff = (tid & 1) * 16;
#pragma unroll
      for (int j = 0; j < 16; ++j) Bst[n][koff + j] = Braw[buf][koff + j][n];
    }
    __syncthreads();
    v16bf a0 = frag_a_raw(&As[buf][0][0], 40, wm * 32, 0, lane);
    v16bf a1 = frag_a_raw(&As[buf][0][0], 40, wm * 32 + 16, 0, lane);
    v16bf b0 = frag_bT(&Bst[0][0], 40, 0, wn * 32, lane);
    v16bf b1 = frag_bT(&Bst[0][0], 40, 0, wn * 32 + 16, lane);
    acc00 = wmma_bf16(a0, b0, acc00);
    acc01 = wmma_bf16(a0, b1, acc01);
    acc10 = wmma_bf16(a1, b0, acc10);
    acc11 = wmma_bf16(a1, b1, acc11);
  }
  const int n = lane & 15, half = lane >> 4;
#pragma unroll
  for (int r = 0; r < 8; ++r) {
    int m0 = mBase + wm * 32 + half * 8 + r;
    int m1 = m0 + 16;
    C[(size_t)m0 * N + nBase + wn * 32 + n]      = acc00[r];
    C[(size_t)m0 * N + nBase + wn * 32 + 16 + n] = acc01[r];
    C[(size_t)m1 * N + nBase + wn * 32 + n]      = acc10[r];
    C[(size_t)m1 * N + nBase + wn * 32 + 16 + n] = acc11[r];
  }
}

// ---------- b/a projections + gates ----------
__global__ __launch_bounds__(256) void ba_kernel(const float* __restrict__ hs,
                                                 const float* __restrict__ Wb,
                                                 const float* __restrict__ Wa,
                                                 const float* __restrict__ A_log,
                                                 const float* __restrict__ dt_bias,
                                                 float* __restrict__ beta,
                                                 float* __restrict__ g) {
  const int t = blockIdx.x;
  __shared__ float row[HDIM];
  for (int i = threadIdx.x; i < HDIM; i += 256) row[i] = hs[(size_t)t * HDIM + i];
  __syncthreads();
  if (threadIdx.x < 64) {
    const int c = threadIdx.x & 31;
    const bool isA = threadIdx.x >= 32;
    const float* W = isA ? Wa : Wb;
    float acc = 0.0f;
    for (int kk = 0; kk < HDIM; ++kk) acc += row[kk] * W[(size_t)kk * HV + c];
    if (!isA) {
      beta[(size_t)t * HV + c] = 1.0f / (1.0f + __expf(-acc));
    } else {
      float x = acc + dt_bias[c];
      float sp = (x > 20.0f) ? x : log1pf(__expf(x));
      g[(size_t)t * HV + c] = -__expf(A_log[c]) * sp;
    }
  }
}

// ---------- causal conv1d (K=4) + SiLU + split into q/k/v ----------
__global__ __launch_bounds__(256) void conv_silu_split_kernel(const float* __restrict__ mixed,
                                                              const float* __restrict__ cw,
                                                              const float* __restrict__ cb,
                                                              float* __restrict__ q,
                                                              float* __restrict__ k,
                                                              float* __restrict__ v) {
  size_t idx = (size_t)blockIdx.x * 256 + threadIdx.x;
  if (idx >= (size_t)S_LEN * CONV_DIM) return;
  int t = (int)(idx / CONV_DIM);
  int c = (int)(idx % CONV_DIM);
  float acc = cb[c];
#pragma unroll
  for (int j = 0; j < KCONV; ++j) {
    int tt = t - (KCONV - 1) + j;
    if (tt >= 0) acc += cw[(size_t)c * KCONV + j] * mixed[(size_t)tt * CONV_DIM + c];
  }
  float o = acc / (1.0f + __expf(-acc));   // silu
  if (c < KEY_DIM)           q[(size_t)t * KEY_DIM + c] = o;
  else if (c < 2 * KEY_DIM)  k[(size_t)t * KEY_DIM + (c - KEY_DIM)] = o;
  else                       v[(size_t)t * VAL_DIM + (c - 2 * KEY_DIM)] = o;
}

// ---------- chunked gated delta rule ----------
// grid: HV * NVB blocks (value-dim blocked: the recurrence is separable per state
// column); block = 256 threads (8 waves). State DK x DVB kept in LDS (f32).
__global__ __launch_bounds__(256) void delta_rule_kernel(const float* __restrict__ qf,
                                                         const float* __restrict__ kf,
                                                         const float* __restrict__ vf,
                                                         const float* __restrict__ gf,
                                                         const float* __restrict__ betaf,
                                                         float* __restrict__ core) {
  extern __shared__ char smem_raw[];
  char* p = smem_raw;
  float*  Sst = (float*)p;  p += DKDIM * DVB * sizeof(float);   // 128x32 state (f32)
  bf16_t* qs  = (bf16_t*)p; p += CHUNK * DKDIM * 2;             // l2norm(q)*DK^-0.5
  bf16_t* ksb = (bf16_t*)p; p += CHUNK * DKDIM * 2;             // l2norm(k)
  bf16_t* wsb = (bf16_t*)p; p += CHUNK * DKDIM * 2;             // w = T @ (k*beta*e^g)
  bf16_t* vs  = (bf16_t*)p; p += CHUNK * DVB * 2;               // v tile
  bf16_t* us  = (bf16_t*)p; p += CHUNK * DVB * 2;               // u = T @ (v*beta)
  bf16_t* vn  = (bf16_t*)p; p += CHUNK * DVB * 2;               // v_new
  bf16_t* Ab  = (bf16_t*)p; p += CHUNK * CHUNK * 2;
  bf16_t* Tb0 = (bf16_t*)p; p += CHUNK * CHUNK * 2;
  bf16_t* Tb1 = (bf16_t*)p; p += CHUNK * CHUNK * 2;
  bf16_t* Pb0 = (bf16_t*)p; p += CHUNK * CHUNK * 2;
  bf16_t* Pb1 = (bf16_t*)p; p += CHUNK * CHUNK * 2;
  bf16_t* atn = (bf16_t*)p; p += CHUNK * CHUNK * 2;
  float* gsh = (float*)p; p += CHUNK * 4;   // cumsum(g) within chunk
  float* bsh = (float*)p; p += CHUNK * 4;   // beta
  float* eg  = (float*)p; p += CHUNK * 4;   // exp(g)
  float* bg  = (float*)p; p += CHUNK * 4;   // beta*exp(g)
  float* es  = (float*)p; p += CHUNK * 4;   // exp(g_last - g)

  const int h    = blockIdx.x >> 2;          // value head
  const int vc   = (blockIdx.x & 3) * DVB;   // value-dim column block
  const int kh   = h >> 1;                   // key head (GQA, rep = HV/HK = 2)
  const int tid  = threadIdx.x;
  const int lane = tid & 31, wave = tid >> 5;

  for (int i = tid; i < DKDIM * DVB; i += 256) Sst[i] = 0.0f;

#pragma unroll 1
  for (int c = 0; c < NCHUNK; ++c) {
    const int t0 = c * CHUNK;
    __syncthreads();   // protect gsh/bsh/es/eg/vn/Sst across chunks
    if (tid < CHUNK) {
      bsh[tid] = betaf[(size_t)(t0 + tid) * HV + h];
      gsh[tid] = gf[(size_t)(t0 + tid) * HV + h];
    }
    __syncthreads();
    if (tid == 0) {   // in-chunk cumulative sum of g
      float s = 0.0f;
      for (int i = 0; i < CHUNK; ++i) { s += gsh[i]; gsh[i] = s; }
    }
    __syncthreads();
    if (tid < CHUNK) {
      float gv = gsh[tid];
      eg[tid] = __expf(gv);
      bg[tid] = bsh[tid] * __expf(gv);
      es[tid] = __expf(gsh[CHUNK - 1] - gv);
    }
    {   // q/k rows with l2norm (4 threads per row, 32 elems each)
      int r = tid >> 2, sub = tid & 3;
      const float* qrow = qf + ((size_t)(t0 + r) * HK + kh) * DKDIM + sub * 32;
      const float* krow = kf + ((size_t)(t0 + r) * HK + kh) * DKDIM + sub * 32;
      float sq = 0.0f, sk = 0.0f;
#pragma unroll
      for (int i = 0; i < 32; ++i) {
        float x = qrow[i]; sq += x * x;
        float y = krow[i]; sk += y * y;
      }
      sq += __shfl_xor(sq, 1); sq += __shfl_xor(sq, 2);
      sk += __shfl_xor(sk, 1); sk += __shfl_xor(sk, 2);
      float qsc = rsqrtf(sq + 1e-6f) * 0.08838834764831845f;  // * DK^-0.5
      float ksc = rsqrtf(sk + 1e-6f);
#pragma unroll
      for (int i = 0; i < 32; ++i) {
        qs[r * DKDIM + sub * 32 + i]  = f2bf(qrow[i] * qsc);
        ksb[r * DKDIM + sub * 32 + i] = f2bf(krow[i] * ksc);
      }
    }
    {   // v tile CHUNK x DVB (8 contiguous per thread)
      int base = tid * 8;
      int r = base >> 5, cc = base & 31;
      const float* vrow = vf + ((size_t)(t0 + r) * HV + h) * DVDIM + vc + cc;
#pragma unroll
      for (int i = 0; i < 8; ++i) vs[base + i] = f2bf(vrow[i]);
    }
    __syncthreads();

    // ---- A = -(k*beta @ k^T) * decay, strictly lower ----  (64x64, K=128)
#pragma unroll
    for (int tI = 0; tI < 2; ++tI) {
      int tile = wave * 2 + tI, tm = tile >> 2, tn = tile & 3;
      v8f acc = {};
#pragma unroll
      for (int kb = 0; kb < 4; ++kb) {
        v16bf af = frag_a_rs(ksb, DKDIM, tm * 16, kb * 32, bsh, lane);
        v16bf bb = frag_bT(ksb, DKDIM, kb * 32, tn * 16, lane);
        acc = wmma_bf16(af, bb, acc);
      }
      int n = lane & 15, half = lane >> 4;
#pragma unroll
      for (int r = 0; r < 8; ++r) {
        int i = tm * 16 + half * 8 + r, j = tn * 16 + n;
        float val = (i > j) ? (-acc[r] * __expf(gsh[i] - gsh[j])) : 0.0f;
        Ab[i * CHUNK + j] = f2bf(val);
      }
    }
    __syncthreads();

    // ---- T = (I - A)^-1 = (I+A)(I+A^2)(I+A^4)(I+A^8)(I+A^16)(I+A^32) (A nilpotent) ----
    for (int i = tid; i < CHUNK * CHUNK; i += 256) {
      int r = i >> 6, j = i & 63;
      Pb0[i] = Ab[i];
      Tb0[i] = f2bf(bf2f(Ab[i]) + ((r == j) ? 1.0f : 0.0f));
    }
    __syncthreads();
    bf16_t *Pc = Pb0, *Pn = Pb1, *Tc = Tb0, *Tn = Tb1;
    for (int it = 0; it < 5; ++it) {
      // Pn = Pc @ Pc
#pragma unroll
      for (int tI = 0; tI < 2; ++tI) {
        int tile = wave * 2 + tI, tm = tile >> 2, tn = tile & 3;
        v8f acc = {};
#pragma unroll
        for (int kb = 0; kb < 2; ++kb) {
          v16bf af = frag_a_raw(Pc, CHUNK, tm * 16, kb * 32, lane);
          v16bf bb = frag_b_raw(Pc, CHUNK, kb * 32, tn * 16, lane);
          acc = wmma_bf16(af, bb, acc);
        }
        store_tile(Pn, CHUNK, tm * 16, tn * 16, acc, lane);
      }
      __syncthreads();
      // Tn = Tc + Pn @ Tc
#pragma unroll
      for (int tI = 0; tI < 2; ++tI) {
        int tile = wave * 2 + tI, tm = tile >> 2, tn = tile & 3;
        v8f acc = {};
#pragma unroll
        for (int kb = 0; kb < 2; ++kb) {
          v16bf af = frag_a_raw(Pn, CHUNK, tm * 16, kb * 32, lane);
          v16bf bb = frag_b_raw(Tc, CHUNK, kb * 32, tn * 16, lane);
          acc = wmma_bf16(af, bb, acc);
        }
        int n = lane & 15, half = lane >> 4;
#pragma unroll
        for (int r = 0; r < 8; ++r) {
          int i = tm * 16 + half * 8 + r, j = tn * 16 + n;
          Tn[i * CHUNK + j] = f2bf(acc[r] + bf2f(Tc[i * CHUNK + j]));
        }
      }
      __syncthreads();
      bf16_t* t;
      t = Pc; Pc = Pn; Pn = t;
      t = Tc; Tc = Tn; Tn = t;
    }

    // ---- u = T @ (v*beta)  (64xDVB)  and  w = T @ (k*beta*e^g)  (64x128) ---- 40 tiles, 5/wave
#pragma unroll
    for (int tI = 0; tI < 5; ++tI) {
      int tile = wave * 5 + tI;
      v8f acc = {};
      if (tile < 8) {
        int tm = tile >> 1, tn = tile & 1;
#pragma unroll
        for (int kb = 0; kb < 2; ++kb) {
          v16bf af = frag_a_raw(Tc, CHUNK, tm * 16, kb * 32, lane);
          v16bf bb = frag_b_ks(vs, DVB, kb * 32, tn * 16, bsh, lane);
          acc = wmma_bf16(af, bb, acc);
        }
        store_tile(us, DVB, tm * 16, tn * 16, acc, lane);
      } else {
        int tl = tile - 8, tm = tl >> 3, tn = tl & 7;
#pragma unroll
        for (int kb = 0; kb < 2; ++kb) {
          v16bf af = frag_a_raw(Tc, CHUNK, tm * 16, kb * 32, lane);
          v16bf bb = frag_b_ks(ksb, DKDIM, kb * 32, tn * 16, bg, lane);
          acc = wmma_bf16(af, bb, acc);
        }
        store_tile(wsb, DKDIM, tm * 16, tn * 16, acc, lane);
      }
    }

    // ---- attn = tril(q @ k^T) * decay ----  (64x64, K=128)
#pragma unroll
    for (int tI = 0; tI < 2; ++tI) {
      int tile = wave * 2 + tI, tm = tile >> 2, tn = tile & 3;
      v8f acc = {};
#pragma unroll
      for (int kb = 0; kb < 4; ++kb) {
        v16bf af = frag_a_raw(qs, DKDIM, tm * 16, kb * 32, lane);
        v16bf bb = frag_bT(ksb, DKDIM, kb * 32, tn * 16, lane);
        acc = wmma_bf16(af, bb, acc);
      }
      int n = lane & 15, half = lane >> 4;
#pragma unroll
      for (int r = 0; r < 8; ++r) {
        int i = tm * 16 + half * 8 + r, j = tn * 16 + n;
        float val = (i >= j) ? (acc[r] * __expf(gsh[i] - gsh[j])) : 0.0f;
        atn[i * CHUNK + j] = f2bf(val);
      }
    }
    __syncthreads();

    // ---- v_new = u - w @ S ----  (64xDVB, K=128): 8 tiles, 1/wave
    {
      int tm = wave >> 1, tn = wave & 1;
      v8f acc = {};
#pragma unroll
      for (int kb = 0; kb < 4; ++kb) {
        v16bf af = frag_a_raw(wsb, DKDIM, tm * 16, kb * 32, lane);
        v16bf bb = frag_b_f32(Sst, DVB, kb * 32, tn * 16, lane);
        acc = wmma_bf16(af, bb, acc);
      }
      int n = lane & 15, half = lane >> 4;
#pragma unroll
      for (int r = 0; r < 8; ++r) {
        int i = tm * 16 + half * 8 + r, j = tn * 16 + n;
        vn[i * DVB + j] = f2bf(bf2f(us[i * DVB + j]) - acc[r]);
      }
    }
    __syncthreads();

    // ---- out = (q*e^g) @ S + attn @ v_new ----  (64xDVB): 8 tiles, 1/wave
    {
      int tm = wave >> 1, tn = wave & 1;
      v8f acc = {};
#pragma unroll
      for (int kb = 0; kb < 4; ++kb) {
        v16bf af = frag_a_rs(qs, DKDIM, tm * 16, kb * 32, eg, lane);
        v16bf bb = frag_b_f32(Sst, DVB, kb * 32, tn * 16, lane);
        acc = wmma_bf16(af, bb, acc);
      }
#pragma unroll
      for (int kb = 0; kb < 2; ++kb) {
        v16bf af = frag_a_raw(atn, CHUNK, tm * 16, kb * 32, lane);
        v16bf bb = frag_b_raw(vn, DVB, kb * 32, tn * 16, lane);
        acc = wmma_bf16(af, bb, acc);
      }
      int n = lane & 15, half = lane >> 4;
#pragma unroll
      for (int r = 0; r < 8; ++r) {
        int i = tm * 16 + half * 8 + r, j = tn * 16 + n;
        core[((size_t)(t0 + i) * HV + h) * DVDIM + vc + j] = acc[r];
      }
    }
    __syncthreads();   // everyone done reading Sst before updating it

    // ---- S = S*e^{gl} + (k * e^{gl-g})^T @ v_new ----  (128xDVB, K=64): 16 tiles, 2/wave
    {
      float egl = __expf(gsh[CHUNK - 1]);
#pragma unroll
      for (int tI = 0; tI < 2; ++tI) {
        int tile = wave * 2 + tI, tm = tile >> 1, tn = tile & 1;
        v8f acc = {};
#pragma unroll
        for (int kb = 0; kb < 2; ++kb) {
          v16bf af = frag_aT(ksb, DKDIM, tm * 16, kb * 32, es, lane);
          v16bf bb = frag_b_raw(vn, DVB, kb * 32, tn * 16, lane);
          acc = wmma_bf16(af, bb, acc);
        }
        int n = lane & 15, half = lane >> 4;
#pragma unroll
        for (int r = 0; r < 8; ++r) {
          int d = tm * 16 + half * 8 + r, e = tn * 16 + n;
          Sst[d * DVB + e] = Sst[d * DVB + e] * egl + acc[r];
        }
      }
    }
  }
}

// ---------- RMSNorm + SiLU(z) gate, write bf16 for the final GEMM ----------
__global__ __launch_bounds__(256) void gate_kernel(const float* __restrict__ core,
                                                   const float* __restrict__ z,
                                                   const float* __restrict__ norm_w,
                                                   bf16_t* __restrict__ gated) {
  int wid = blockIdx.x * 8 + (threadIdx.x >> 5);   // one wave per (t, h)
  int lane = threadIdx.x & 31;
  int t = wid >> 5, h = wid & 31;
  const float* cr = core + ((size_t)t * HV + h) * DVDIM;
  const float* zr = z + ((size_t)t * HV + h) * DVDIM;
  float ss = 0.0f;
  float vals[4];
#pragma unroll
  for (int i = 0; i < 4; ++i) { float x = cr[lane + 32 * i]; vals[i] = x; ss += x * x; }
#pragma unroll
  for (int off = 16; off > 0; off >>= 1) ss += __shfl_xor(ss, off);
  float inv = rsqrtf(ss * (1.0f / DVDIM) + 1e-6f);
#pragma unroll
  for (int i = 0; i < 4; ++i) {
    int d = lane + 32 * i;
    float zz = zr[d];
    float o = vals[i] * inv * norm_w[d] * (zz / (1.0f + __expf(-zz)));
    gated[((size_t)t * HV + h) * DVDIM + d] = f2bf(o);
  }
}

// ---------- host launcher ----------
extern "C" void kernel_launch(void* const* d_in, const int* in_sizes, int n_in,
                              void* d_out, int out_size, void* d_ws, size_t ws_size,
                              hipStream_t stream) {
  (void)in_sizes; (void)n_in; (void)out_size; (void)ws_size;
  const float* hs      = (const float*)d_in[0];
  const float* Wqkv    = (const float*)d_in[1];
  const float* Wz      = (const float*)d_in[2];
  const float* Wb      = (const float*)d_in[3];
  const float* Wa      = (const float*)d_in[4];
  const float* convw   = (const float*)d_in[5];
  const float* convb   = (const float*)d_in[6];
  const float* A_log   = (const float*)d_in[7];
  const float* dt_bias = (const float*)d_in[8];
  const float* norm_w  = (const float*)d_in[9];
  const float* Wout    = (const float*)d_in[10];
  float* out = (float*)d_out;

  // workspace carve-up
  char* w = (char*)d_ws;
  size_t off = 0;
  auto take = [&](size_t bytes) -> void* {
    void* r = w + off;
    off = (off + bytes + 255) & ~(size_t)255;
    return r;
  };
  bf16_t* hs_bf   = (bf16_t*)take((size_t)S_LEN * HDIM * 2);
  bf16_t* wqkv_bf = (bf16_t*)take((size_t)HDIM * CONV_DIM * 2);
  bf16_t* wz_bf   = (bf16_t*)take((size_t)HDIM * VAL_DIM * 2);
  bf16_t* wout_bf = (bf16_t*)take((size_t)VAL_DIM * HDIM * 2);
  float*  mixed   = (float*)take((size_t)S_LEN * CONV_DIM * 4);   // reused: core + gated
  float*  zbuf    = (float*)take((size_t)S_LEN * VAL_DIM * 4);
  float*  qbuf    = (float*)take((size_t)S_LEN * KEY_DIM * 4);
  float*  kbuf    = (float*)take((size_t)S_LEN * KEY_DIM * 4);
  float*  vbuf    = (float*)take((size_t)S_LEN * VAL_DIM * 4);
  float*  betabuf = (float*)take((size_t)S_LEN * HV * 4);
  float*  gbuf    = (float*)take((size_t)S_LEN * HV * 4);
  // mixed (128MB) is dead after the conv; reuse it for core (64MB f32) + gated (32MB bf16)
  float*  corebuf = mixed;
  bf16_t* gated   = (bf16_t*)((char*)mixed + (size_t)S_LEN * VAL_DIM * 4);

  // 1) pack weights + activations to bf16
  {
    size_t n;
    n = (size_t)S_LEN * HDIM;
    cvt_bf16_kernel<<<dim3((unsigned)((n + 255) / 256)), dim3(256), 0, stream>>>(hs, hs_bf, n);
    n = (size_t)HDIM * CONV_DIM;
    cvt_bf16_kernel<<<dim3((unsigned)((n + 255) / 256)), dim3(256), 0, stream>>>(Wqkv, wqkv_bf, n);
    n = (size_t)HDIM * VAL_DIM;
    cvt_bf16_kernel<<<dim3((unsigned)((n + 255) / 256)), dim3(256), 0, stream>>>(Wz, wz_bf, n);
    n = (size_t)VAL_DIM * HDIM;
    cvt_bf16_kernel<<<dim3((unsigned)((n + 255) / 256)), dim3(256), 0, stream>>>(Wout, wout_bf, n);
  }
  // 2) mixed = hs @ W_qkv   (4096x2048 @ 2048x8192)
  wmma_gemm_kernel<<<dim3(CONV_DIM / 128, S_LEN / 64), dim3(256), 0, stream>>>(
      hs_bf, wqkv_bf, mixed, S_LEN, CONV_DIM, HDIM);
  // 3) z = hs @ W_z         (4096x2048 @ 2048x4096)
  wmma_gemm_kernel<<<dim3(VAL_DIM / 128, S_LEN / 64), dim3(256), 0, stream>>>(
      hs_bf, wz_bf, zbuf, S_LEN, VAL_DIM, HDIM);
  // 4) beta / g
  ba_kernel<<<dim3(S_LEN), dim3(256), 0, stream>>>(hs, Wb, Wa, A_log, dt_bias, betabuf, gbuf);
  // 5) causal conv1d + silu + split
  {
    size_t n = (size_t)S_LEN * CONV_DIM;
    conv_silu_split_kernel<<<dim3((unsigned)((n + 255) / 256)), dim3(256), 0, stream>>>(
        mixed, convw, convb, qbuf, kbuf, vbuf);
  }
  // 6) chunked gated delta rule
  {
    size_t smem = (size_t)DKDIM * DVB * 4       // Sst
                + 3 * (size_t)CHUNK * DKDIM * 2 // qs, ksb, wsb
                + 3 * (size_t)CHUNK * DVB * 2   // vs, us, vn
                + 6 * (size_t)CHUNK * CHUNK * 2 // Ab, Tb0, Tb1, Pb0, Pb1, atn
                + 5 * (size_t)CHUNK * 4;        // gsh, bsh, eg, bg, es
    delta_rule_kernel<<<dim3(HV * NVB), dim3(256), smem, stream>>>(
        qbuf, kbuf, vbuf, gbuf, betabuf, corebuf);
  }
  // 7) RMSNorm + SiLU(z) gating -> bf16
  gate_kernel<<<dim3(S_LEN * HV / 8), dim3(256), 0, stream>>>(corebuf, zbuf, norm_w, gated);
  // 8) out = gated @ W_out  (4096x4096 @ 4096x2048)
  wmma_gemm_kernel<<<dim3(HDIM / 128, S_LEN / 64), dim3(256), 0, stream>>>(
      gated, wout_bf, out, S_LEN, HDIM, VAL_DIM);
}
